// TransformerBlock_30812095381604
// MI455X (gfx1250) — matrix operand order repaired and checked by reading, not compiled, not run on hardware
//
#include <hip/hip_runtime.h>

typedef __attribute__((ext_vector_type(16))) __bf16 v16bf;
typedef __attribute__((ext_vector_type(8)))  __bf16 v8bf;
typedef __attribute__((ext_vector_type(4)))  __bf16 v4bf;
typedef __attribute__((ext_vector_type(8)))  float  v8f;
typedef __attribute__((ext_vector_type(4)))  float  v4f;

// --- Optional CDNA5 async global->LDS copy path (ASYNCcnt) ------------------
#if defined(__has_builtin)
#if __has_builtin(__builtin_amdgcn_global_load_async_to_lds_b128) && \
    __has_builtin(__builtin_amdgcn_s_wait_asynccnt)
#define HAVE_ASYNC_LDS 1
#endif
#endif
#ifndef HAVE_ASYNC_LDS
#define HAVE_ASYNC_LDS 0
#endif

#if HAVE_ASYNC_LDS
typedef int v4i_128 __attribute__((vector_size(16)));
typedef __attribute__((address_space(1))) v4i_128* gv4i_p;  // global int4*
typedef __attribute__((address_space(3))) v4i_128* lv4i_p;  // LDS int4*
static __device__ __forceinline__ void async_cp16(const void* g, void* l) {
  __builtin_amdgcn_global_load_async_to_lds_b128((gv4i_p)(void*)g, (lv4i_p)l,
                                                 0, 0);
}
static __device__ __forceinline__ void async_wait0() {
  __builtin_amdgcn_s_wait_asynccnt(0);
}
#endif

static __device__ __forceinline__ v8f zero8() {
  v8f z;
#pragma unroll
  for (int i = 0; i < 8; ++i) z[i] = 0.0f;
  return z;
}

static __device__ __forceinline__ v16bf pack16(v8bf lo, v8bf hi) {
  v16bf r;
#pragma unroll
  for (int i = 0; i < 8; ++i) { r[i] = lo[i]; r[i + 8] = hi[i]; }
  return r;
}

static __device__ __forceinline__ v8f wmma_bf16(v16bf a, v16bf b, v8f c) {
  // D = A(16x32 bf16) * B(32x16 bf16) + C(16x16 f32)
  return __builtin_amdgcn_wmma_f32_16x16x32_bf16(false, a, false, b, (short)0, c,
                                                 false, false);
}

// ---------------------------------------------------------------------------
// Elementwise f32 -> bf16 (weights / x converted once per launch)
// ---------------------------------------------------------------------------
__global__ void cvt_f32_bf16(const float* __restrict__ s, __bf16* __restrict__ d,
                             int n4) {
  int i = blockIdx.x * blockDim.x + threadIdx.x;
  if (i < n4) {
    v4f a = *(const v4f*)(s + (size_t)i * 4);
    v4bf b;
    b[0] = (__bf16)a[0]; b[1] = (__bf16)a[1];
    b[2] = (__bf16)a[2]; b[3] = (__bf16)a[3];
    *(v4bf*)(d + (size_t)i * 4) = b;
  }
}

// ---------------------------------------------------------------------------
// Tiled GEMM (bf16 in, f32 accumulate):
//   out = act((A[M,K] @ W[K,N] + bias[N]) * outscale) (+ R[M,N])
// Block: 256 threads (8 waves), tile 128x128, K-step 32; each wave owns a
// 32x64 micro tile = 8 v_wmma per K-step. A-tile staged with async copies
// (layout-preserving); B-tile transposed via VGPRs. WF32/WBF16 select which
// output copies are written (branch-free epilogue).
// ---------------------------------------------------------------------------
#define GT_M 128
#define GT_N 128
#define GT_K 32

template <int RELU, int HAS_RES, int WF32, int WBF16>
__global__ __launch_bounds__(256, 2) void gemm_bf16_kernel(
    const __bf16* __restrict__ A, const __bf16* __restrict__ W,
    const float* __restrict__ bias, const float* __restrict__ R,
    float* __restrict__ Cf, __bf16* __restrict__ Cb, int M, int N, int K,
    float outscale) {
  __shared__ __align__(16) __bf16 As[GT_M * GT_K];  // [m][k]
  __shared__ __align__(16) __bf16 Bs[GT_N * GT_K];  // [n][k]  (transposed)

  const int t    = threadIdx.x;
  const int wv   = t >> 5;
  const int lane = t & 31;
  const int l16  = lane & 15;
  const int hf   = lane >> 4;

  const int m0 = blockIdx.x * GT_M;
  const int n0 = blockIdx.y * GT_N;
  const int wm = (wv >> 1) * 32;  // 0,32,64,96
  const int wn = (wv & 1) * 64;   // 0,64

  v8f acc[2][4];
#pragma unroll
  for (int i = 0; i < 2; ++i)
#pragma unroll
    for (int j = 0; j < 4; ++j) acc[i][j] = zero8();

  for (int k0 = 0; k0 < K; k0 += GT_K) {
    __syncthreads();
    // Prefetch next K-tile (global_prefetch_b8)
    if (k0 + GT_K < K) {
      __builtin_prefetch(A + (size_t)(m0 + (t >> 1)) * K + k0 + GT_K + (t & 1) * 16, 0, 1);
      __builtin_prefetch(W + (size_t)(k0 + GT_K + (t >> 3)) * N + n0 + (t & 7) * 16, 0, 1);
    }
    // Stage A tile 128x32 bf16 (8KB): layout-preserving byte copy
#pragma unroll
    for (int j = 0; j < 2; ++j) {
      int idx = t + j * 256;  // 512 x 16B segments (4 per 64B row)
      int row = idx >> 2;
      int seg = idx & 3;
      const __bf16* src = A + (size_t)(m0 + row) * K + k0 + seg * 8;
      __bf16* dst = As + row * GT_K + seg * 8;
#if HAVE_ASYNC_LDS
      async_cp16(src, dst);
#else
      *(v8bf*)dst = *(const v8bf*)src;
#endif
    }
    // Stage B tile transposed: W[k][n] -> Bs[n][k]
#pragma unroll
    for (int j = 0; j < 2; ++j) {
      int idx = t + j * 256;  // 512 x v8bf (16 per 128-elem row)
      int kr  = idx >> 4;
      int seg = idx & 15;
      v8bf w8 = *(const v8bf*)(W + (size_t)(k0 + kr) * N + n0 + seg * 8);
#pragma unroll
      for (int c = 0; c < 8; ++c) Bs[(seg * 8 + c) * GT_K + kr] = w8[c];
    }
#if HAVE_ASYNC_LDS
    async_wait0();
#endif
    __syncthreads();

    // Fragments (contiguous 16B ds_load_b128 pairs, per ISA 16-bit layouts)
    v16bf af[2], bfr[4];
#pragma unroll
    for (int i = 0; i < 2; ++i) {
      const __bf16* p = As + (wm + i * 16 + l16) * GT_K;
      af[i] = pack16(*(const v8bf*)(p + hf * 8),
                     *(const v8bf*)(p + 16 + hf * 8));
    }
#pragma unroll
    for (int j = 0; j < 4; ++j) {
      const __bf16* p = Bs + (wn + j * 16 + l16) * GT_K;
      bfr[j] = pack16(*(const v8bf*)(p + hf * 8),
                      *(const v8bf*)(p + 16 + hf * 8));
    }
#pragma unroll
    for (int i = 0; i < 2; ++i)
#pragma unroll
      for (int j = 0; j < 4; ++j) acc[i][j] = wmma_bf16(af[i], bfr[j], acc[i][j]);
  }

  // Branch-free epilogue
#pragma unroll
  for (int j = 0; j < 4; ++j) {
    int n = n0 + wn + j * 16 + l16;
    float bv = bias[n];
#pragma unroll
    for (int i = 0; i < 2; ++i) {
#pragma unroll
      for (int r = 0; r < 8; ++r) {
        int m = m0 + wm + i * 16 + r + 8 * hf;  // C layout: VGPR r, half hf
        float v = (acc[i][j][r] + bv) * outscale;
        if (RELU) v = fmaxf(v, 0.0f);
        if (HAS_RES) v += R[(size_t)m * N + n];
        if (WF32)  Cf[(size_t)m * N + n] = v;
        if (WBF16) Cb[(size_t)m * N + n] = (__bf16)v;
      }
    }
  }
}

// ---------------------------------------------------------------------------
// Flash attention over bf16 Q/K/V (scale pre-folded into Q by the q-GEMM).
// Block = (b, h, 64 q-rows), 128 threads = 4 waves, 16 q-rows per wave.
// K tile staged with async copies (layout-preserving); V transposed via
// VGPRs. Row sums of exp(P) via WMMA against a ones fragment.
// ---------------------------------------------------------------------------
#define A_S 2048
#define A_D 1024
#define A_HD 64

__global__ __launch_bounds__(128, 2) void attn_kernel(
    const __bf16* __restrict__ Q, const __bf16* __restrict__ K,
    const __bf16* __restrict__ V, __bf16* __restrict__ O) {
  __shared__ __align__(16) __bf16 Kt[32 * 64];       // [k_row][hd]
  __shared__ __align__(16) __bf16 Vt[64 * 32];       // [hd][k_row] (transposed)
  __shared__ __align__(16) __bf16 Pb[4 * 16 * 32];   // per-wave P scratch

  const int t    = threadIdx.x;
  const int wv   = t >> 5;
  const int lane = t & 31;
  const int l16  = lane & 15;
  const int hf   = lane >> 4;

  const int bh = blockIdx.y;
  const int b  = bh >> 4;
  const int h  = bh & 15;
  const int qbase = blockIdx.x * 64 + wv * 16;

  const __bf16* Qb = Q + (size_t)b * A_S * A_D + h * A_HD;
  const __bf16* Kb = K + (size_t)b * A_S * A_D + h * A_HD;
  const __bf16* Vb = V + (size_t)b * A_S * A_D + h * A_HD;

  // Resident Q fragments (A layout); scale already folded in by q-GEMM
  v16bf qf[2];
  {
    const __bf16* qp = Qb + (size_t)(qbase + l16) * A_D;
#pragma unroll
    for (int f = 0; f < 2; ++f)
      qf[f] = pack16(*(const v8bf*)(qp + f * 32 + hf * 8),
                     *(const v8bf*)(qp + f * 32 + 16 + hf * 8));
  }

  // Ones fragment for WMMA-based row sums
  v16bf ones;
#pragma unroll
  for (int i = 0; i < 16; ++i) ones[i] = (__bf16)1.0f;

  float mst[8], lst[8], alph[8];
  v8f accO[4];
#pragma unroll
  for (int r = 0; r < 8; ++r) { mst[r] = -3.0e38f; lst[r] = 0.0f; }
#pragma unroll
  for (int g = 0; g < 4; ++g) accO[g] = zero8();

  __bf16* myP = Pb + wv * (16 * 32);

  for (int kt = 0; kt < A_S / 32; ++kt) {
    __syncthreads();
    // Stage K tile 32x64 bf16 (4KB): layout-preserving byte copy
#pragma unroll
    for (int j = 0; j < 2; ++j) {
      int idx = t + j * 128;  // 256 x 16B segments (4 per 128B row)
      int row = idx >> 2;
      int seg = idx & 3;
      const __bf16* src = Kb + (size_t)(kt * 32 + row) * A_D + seg * 8;
      __bf16* dst = Kt + row * 64 + seg * 8;
#if HAVE_ASYNC_LDS
      async_cp16(src, dst);
#else
      *(v8bf*)dst = *(const v8bf*)src;
#endif
    }
    // Stage V tile transposed: V[k_row][hd] -> Vt[hd][k_row]
#pragma unroll
    for (int j = 0; j < 2; ++j) {
      int idx = t + j * 128;  // 256 x v8bf (8 per 64-elem row)
      int row = idx >> 3;
      int seg = idx & 7;
      v8bf vx = *(const v8bf*)(Vb + (size_t)(kt * 32 + row) * A_D + seg * 8);
#pragma unroll
      for (int c = 0; c < 8; ++c) Vt[(seg * 8 + c) * 32 + row] = vx[c];
    }
#if HAVE_ASYNC_LDS
    async_wait0();
#endif
    __syncthreads();

    // Scores: two 16x16 key-column tiles, contracting head dim 64
    v8f sc[2];
#pragma unroll
    for (int g = 0; g < 2; ++g) {
      sc[g] = zero8();
#pragma unroll
      for (int d = 0; d < 2; ++d) {
        const __bf16* p = Kt + (g * 16 + l16) * 64 + d * 32;
        v16bf kb = pack16(*(const v8bf*)(p + hf * 8),
                          *(const v8bf*)(p + 16 + hf * 8));
        sc[g] = wmma_bf16(qf[d], kb, sc[g]);
      }
    }

    // Online softmax; rows live at (VGPR r, half hf); columns = 16 lanes
#pragma unroll
    for (int r = 0; r < 8; ++r) {
      float s0 = sc[0][r], s1 = sc[1][r];
      float mt = fmaxf(s0, s1);
#pragma unroll
      for (int off = 1; off < 16; off <<= 1)
        mt = fmaxf(mt, __shfl_xor(mt, off, 32));
      float mn    = fmaxf(mst[r], mt);
      float alpha = __expf(mst[r] - mn);
      float p0 = __expf(s0 - mn);
      float p1 = __expf(s1 - mn);
      alph[r] = alpha;
      mst[r]  = mn;
#pragma unroll
      for (int g = 0; g < 4; ++g) accO[g][r] = accO[g][r] * alpha;
      int prow = r + 8 * hf;
      myP[prow * 32 + l16]      = (__bf16)p0;   // C layout -> LDS
      myP[prow * 32 + 16 + l16] = (__bf16)p1;
    }
    __syncthreads();

    // Re-stripe P into A layout; row sums via WMMA against ones; then P @ V
    const __bf16* pp = myP + l16 * 32;
    v16bf pf = pack16(*(const v8bf*)(pp + hf * 8),
                      *(const v8bf*)(pp + 16 + hf * 8));
    v8f rs8 = wmma_bf16(pf, ones, zero8());
#pragma unroll
    for (int r = 0; r < 8; ++r) lst[r] = lst[r] * alph[r] + rs8[r];
#pragma unroll
    for (int g = 0; g < 4; ++g) {
      const __bf16* vp = Vt + (g * 16 + l16) * 32;
      v16bf vb = pack16(*(const v8bf*)(vp + hf * 8),
                        *(const v8bf*)(vp + 16 + hf * 8));
      accO[g] = wmma_bf16(pf, vb, accO[g]);
    }
  }

  // Normalize and write ctx (bf16) at [b, q, h*64 + hd]
  __bf16* Ob = O + (size_t)b * A_S * A_D + h * A_HD;
#pragma unroll
  for (int r = 0; r < 8; ++r) {
    float inv = 1.0f / lst[r];
    int qrow  = qbase + r + 8 * hf;
#pragma unroll
    for (int g = 0; g < 4; ++g)
      Ob[(size_t)qrow * A_D + g * 16 + l16] = (__bf16)(accO[g][r] * inv);
  }
}

// ---------------------------------------------------------------------------
extern "C" void kernel_launch(void* const* d_in, const int* in_sizes, int n_in,
                              void* d_out, int out_size, void* d_ws,
                              size_t ws_size, hipStream_t stream) {
  (void)in_sizes; (void)n_in; (void)out_size; (void)ws_size;
  const float* x  = (const float*)d_in[0];
  const float* We = (const float*)d_in[1];
  const float* be = (const float*)d_in[2];
  const float* Wq = (const float*)d_in[3];
  const float* bq = (const float*)d_in[4];
  const float* Wk = (const float*)d_in[5];
  const float* bk = (const float*)d_in[6];
  const float* Wv = (const float*)d_in[7];
  const float* bv = (const float*)d_in[8];
  const float* W0 = (const float*)d_in[9];
  const float* b0 = (const float*)d_in[10];
  const float* W1 = (const float*)d_in[11];
  const float* b1 = (const float*)d_in[12];
  const float* W2 = (const float*)d_in[13];
  const float* b2 = (const float*)d_in[14];

  const int M = 2 * 2048;  // B*S
  const int D = 1024, DIN = 512;
  const size_t mat = (size_t)M * D;

  // Workspace carve-up (bf16 chain; f32 only for residual sources)
  char* w = (char*)d_ws;
  __bf16* xb   = (__bf16*)w; w += (size_t)M * DIN * 2;
  __bf16* Web  = (__bf16*)w; w += (size_t)DIN * D * 2;
  __bf16* Wqb  = (__bf16*)w; w += (size_t)D * D * 2;
  __bf16* Wkb  = (__bf16*)w; w += (size_t)D * D * 2;
  __bf16* Wvb  = (__bf16*)w; w += (size_t)D * D * 2;
  __bf16* W0b  = (__bf16*)w; w += (size_t)D * D * 2;
  __bf16* W1b  = (__bf16*)w; w += (size_t)D * D * 2;
  __bf16* W2b  = (__bf16*)w; w += (size_t)D * D * 2;
  float*  e    = (float*)w;  w += mat * 4;
  __bf16* eb   = (__bf16*)w; w += mat * 2;
  __bf16* qb   = (__bf16*)w; w += mat * 2;
  __bf16* kb   = (__bf16*)w; w += mat * 2;
  __bf16* vb   = (__bf16*)w; w += mat * 2;
  __bf16* ctxb = (__bf16*)w; w += mat * 2;
  float*  mha  = (float*)w;  w += mat * 4;
  __bf16* mhab = (__bf16*)w; w += mat * 2;
  __bf16* ffb  = (__bf16*)w; w += mat * 2;

  // One-time (per launch) f32 -> bf16 conversion of x and weights
  auto cvt = [&](const float* s, __bf16* d, size_t n) {
    int n4 = (int)(n / 4);
    cvt_f32_bf16<<<(n4 + 255) / 256, 256, 0, stream>>>(s, d, n4);
  };
  cvt(x,  xb,  (size_t)M * DIN);
  cvt(We, Web, (size_t)DIN * D);
  cvt(Wq, Wqb, (size_t)D * D);
  cvt(Wk, Wkb, (size_t)D * D);
  cvt(Wv, Wvb, (size_t)D * D);
  cvt(W0, W0b, (size_t)D * D);
  cvt(W1, W1b, (size_t)D * D);
  cvt(W2, W2b, (size_t)D * D);

  dim3 gg(M / GT_M, D / GT_N);
  // e = x@We + be            (f32 for residual, bf16 for next GEMMs)
  gemm_bf16_kernel<0, 0, 1, 1><<<gg, 256, 0, stream>>>(
      xb, Web, be, nullptr, e, eb, M, D, DIN, 1.0f);
  // q/k/v (q pre-scaled by 1/sqrt(64))
  gemm_bf16_kernel<0, 0, 0, 1><<<gg, 256, 0, stream>>>(
      eb, Wqb, bq, nullptr, nullptr, qb, M, D, D, 0.125f);
  gemm_bf16_kernel<0, 0, 0, 1><<<gg, 256, 0, stream>>>(
      eb, Wkb, bk, nullptr, nullptr, kb, M, D, D, 1.0f);
  gemm_bf16_kernel<0, 0, 0, 1><<<gg, 256, 0, stream>>>(
      eb, Wvb, bv, nullptr, nullptr, vb, M, D, D, 1.0f);
  // ctx = softmax(q k^T) v   (bf16)
  attn_kernel<<<dim3(A_S / 64, 2 * 16), 128, 0, stream>>>(qb, kb, vb, ctxb);
  // mha = e + ctx@W0 + b0    (f32 for residual, bf16 for FFN)
  gemm_bf16_kernel<0, 1, 1, 1><<<gg, 256, 0, stream>>>(
      ctxb, W0b, b0, e, mha, mhab, M, D, D, 1.0f);
  // ff = relu(mha@W1 + b1)   (bf16 only)
  gemm_bf16_kernel<1, 0, 0, 1><<<gg, 256, 0, stream>>>(
      mhab, W1b, b1, nullptr, nullptr, ffb, M, D, D, 1.0f);
  // out = mha + ff@W2 + b2   (f32 -> d_out)
  gemm_bf16_kernel<0, 1, 1, 0><<<gg, 256, 0, stream>>>(
      ffb, W2b, b2, mha, (float*)d_out, nullptr, M, D, D, 1.0f);
}
